// Encoder_60043642798093
// MI455X (gfx1250) — matrix operand order repaired and checked by reading, not compiled, not run on hardware
//
#include <hip/hip_runtime.h>

#define HD   1024
#define SEQL 4096
#define H3   3072
#define ASTR 40               // padded LDS row stride in bf16 elems (80 B)
#define TILE_K 32
#define BS_OFF (128 * ASTR)   // B region offset inside a slab buffer (elems)
#define SLAB   (256 * ASTR)   // one double-buffer slab (A + B regions, elems)

typedef __attribute__((ext_vector_type(16))) __bf16 v16bf;
typedef __attribute__((ext_vector_type(8)))  __bf16 v8bf;
typedef __attribute__((ext_vector_type(8)))  float  v8f;
typedef __attribute__((ext_vector_type(4)))  int    v4i;

#if __has_builtin(__builtin_amdgcn_global_load_async_to_lds_b128) && \
    __has_builtin(__builtin_amdgcn_s_wait_asynccnt)
#define HAVE_ASYNC 1
#define GLD_ASYNC(gp, lp)                                                  \
    __builtin_amdgcn_global_load_async_to_lds_b128(                        \
        (__attribute__((address_space(1))) v4i*)(gp),                      \
        (__attribute__((address_space(3))) v4i*)(lp), 0, 0)
#define WAIT_ASYNC(n) __builtin_amdgcn_s_wait_asynccnt(n)
#else
#define HAVE_ASYNC 0
#define WAIT_ASYNC(n)
#endif

// ---------------- Kernel 1: embedding gather + fp32 -> bf16 ----------------
__global__ void embed_to_bf16(const int* __restrict__ tokens,
                              const float* __restrict__ emb,
                              __bf16* __restrict__ xb) {
    int s = blockIdx.x;
    int tok = tokens[s];
    const float* src = emb + (size_t)tok * HD;
    __bf16* dst = xb + (size_t)s * HD;
    for (int i = threadIdx.x; i < HD; i += blockDim.x)
        dst[i] = (__bf16)src[i];
}

// ---------------- Kernel 2: weight fp32 -> bf16 ----------------
__global__ void conv_to_bf16(const float* __restrict__ src,
                             __bf16* __restrict__ dst, int n) {
    int i = blockIdx.x * blockDim.x + threadIdx.x;
    if (i < n) dst[i] = (__bf16)src[i];
}

// ---------------- Kernel 3: gi = X @ W_ih^T + b_ih via WMMA bf16 ----------
// Block tile 128M x 128N, 8 waves as 2(M) x 4(N); wave tile 64M x 32N
// (8 named accumulators). K staged through LDS in 32-wide slabs with async
// global->LDS copies (SADDR form), double buffered, single loop body.
__global__ void __launch_bounds__(256)
gemm_gi_wmma(const __bf16* __restrict__ X,   // SEQL x HD
             const __bf16* __restrict__ W,   // H3 x HD (row n = B column n)
             const float*  __restrict__ bias,
             float* __restrict__ gi) {       // SEQL x H3
    __shared__ __bf16 smem[2][SLAB];         // [buf][ A(128*ASTR) | B(128*ASTR) ]

    const int tid  = threadIdx.x;
    const int lane = tid & 31;
    const int wave = tid >> 5;
    const int wm   = wave >> 2;      // 0..1  (M direction)
    const int wn   = wave & 3;       // 0..3  (N direction)
    const int half = lane >> 4;      // half-wave K select
    const int mn   = lane & 15;      // M row (A) / N col (B,D)
    const int blockN = blockIdx.x * 128;
    const int blockM = blockIdx.y * 128;

    // Loop-invariant staging: i=0,1 stage A (uniform base X), i=2,3 stage B
    // (uniform base W); per-thread state is one 32-bit elem offset + LDS ptr.
    int     goff[4];
    __bf16* lb[4];
    #pragma unroll
    for (int i = 0; i < 4; ++i) {
        int idx = tid + i * 256;          // 0..1023
        int isB = idx >> 9;               // i<2 -> A, i>=2 -> B
        int r   = (idx & 511) >> 2;       // row 0..127
        int seg = idx & 3;                // 8-element segment
        goff[i] = (isB ? (blockN + r) : (blockM + r)) * HD + seg * 8;
        lb[i]   = &smem[0][0] + isB * BS_OFF + r * ASTR + seg * 8;
    }

    auto issue = [&](int buf, int k0) {
#if HAVE_ASYNC
        GLD_ASYNC(X + goff[0] + k0, lb[0] + buf * SLAB);
        GLD_ASYNC(X + goff[1] + k0, lb[1] + buf * SLAB);
        GLD_ASYNC(W + goff[2] + k0, lb[2] + buf * SLAB);
        GLD_ASYNC(W + goff[3] + k0, lb[3] + buf * SLAB);
#else
        *(v8bf*)(lb[0] + buf * SLAB) = *(const v8bf*)(X + goff[0] + k0);
        *(v8bf*)(lb[1] + buf * SLAB) = *(const v8bf*)(X + goff[1] + k0);
        *(v8bf*)(lb[2] + buf * SLAB) = *(const v8bf*)(W + goff[2] + k0);
        *(v8bf*)(lb[3] + buf * SLAB) = *(const v8bf*)(W + goff[3] + k0);
#endif
    };

    // Eight named accumulators -> in-place WMMA accumulate.
    v8f a00 = {}, a01 = {}, a10 = {}, a11 = {};
    v8f a20 = {}, a21 = {}, a30 = {}, a31 = {};

#define MT_STEP(mt, accL, accR)                                              \
    do {                                                                     \
        const __bf16* ap = A0 + (wm * 64 + (mt) * 16 + mn) * ASTR;           \
        v8bf alo = *(const v8bf*)(ap + half * 8);                            \
        v8bf ahi = *(const v8bf*)(ap + 16 + half * 8);                       \
        v16bf af = __builtin_shufflevector(alo, ahi,                         \
                       0,1,2,3,4,5,6,7,8,9,10,11,12,13,14,15);               \
        accL = __builtin_amdgcn_wmma_f32_16x16x32_bf16(                      \
                   false, af, false, bfrag0, (short)0, accL, false, false);  \
        accR = __builtin_amdgcn_wmma_f32_16x16x32_bf16(                      \
                   false, af, false, bfrag1, (short)0, accR, false, false);  \
    } while (0)

    auto compute = [&](int buf) {
        const __bf16* A0 = &smem[buf][0];
        const __bf16* B0 = &smem[buf][BS_OFF];
        // B fragments: lane = column (W row), elems 0-7 K=half*8+e, 8-15 K=16+half*8+(e-8)
        const __bf16* bp0 = B0 + (wn * 32 + mn) * ASTR + half * 8;
        const __bf16* bp1 = bp0 + 16 * ASTR;
        v8bf b0lo = *(const v8bf*)bp0;
        v8bf b0hi = *(const v8bf*)(bp0 + 16);
        v8bf b1lo = *(const v8bf*)bp1;
        v8bf b1hi = *(const v8bf*)(bp1 + 16);
        v16bf bfrag0 = __builtin_shufflevector(b0lo, b0hi,
                           0,1,2,3,4,5,6,7,8,9,10,11,12,13,14,15);
        v16bf bfrag1 = __builtin_shufflevector(b1lo, b1hi,
                           0,1,2,3,4,5,6,7,8,9,10,11,12,13,14,15);
        // A fragments: lane = row, elems 0-7 K=half*8+e, 8-15 K=16+half*8+(e-8)
        MT_STEP(0, a00, a01);
        MT_STEP(1, a10, a11);
        MT_STEP(2, a20, a21);
        MT_STEP(3, a30, a31);
    };

    // Pipelined main loop (single body), final slab peeled.
    issue(0, 0);
    int buf = 0;
    #pragma clang loop unroll(disable)
    for (int k0 = 0; k0 < HD - TILE_K; k0 += TILE_K) {
        issue(buf ^ 1, k0 + TILE_K);
        WAIT_ASYNC(4);             // current slab's 4 transfers done
        __syncthreads();
        compute(buf);
        __syncthreads();           // protect buf before next slab overwrites
        buf ^= 1;
    }
    WAIT_ASYNC(0);
    __syncthreads();
    compute(buf);

    // Epilogue: D lane column = mn, VGPR r holds row M = r + 8*half
#define STORE_ACC(mt, nt, accv)                                              \
    do {                                                                     \
        int col = blockN + wn * 32 + (nt) * 16 + mn;                         \
        float bv = bias[col];                                                \
        float* cp = gi + (size_t)(blockM + wm * 64 + (mt) * 16) * H3 + col;  \
        _Pragma("unroll")                                                    \
        for (int r = 0; r < 8; ++r)                                          \
            cp[(size_t)(r + 8 * half) * H3] = accv[r] + bv;                  \
    } while (0)

    STORE_ACC(0, 0, a00); STORE_ACC(0, 1, a01);
    STORE_ACC(1, 0, a10); STORE_ACC(1, 1, a11);
    STORE_ACC(2, 0, a20); STORE_ACC(2, 1, a21);
    STORE_ACC(3, 0, a30); STORE_ACC(3, 1, a31);
#undef MT_STEP
#undef STORE_ACC
}

// ---------------- Kernel 4: zero the h double-buffer ----------------
__global__ void zero_h(float* __restrict__ hbuf) {
    int i = blockIdx.x * blockDim.x + threadIdx.x;
    if (i < 2 * HD) hbuf[i] = 0.0f;
}

// ---------------- Kernel 5: one GRU timestep ----------------
__global__ void gru_step(const __bf16* __restrict__ Whh,   // H3 x HD bf16
                         const float*  __restrict__ bhh,   // H3
                         const float*  __restrict__ gi_t,  // H3 (row t)
                         const float*  __restrict__ h_prev,// HD
                         float* __restrict__ h_next,       // HD
                         float* __restrict__ out_t) {      // HD
    const int lane = threadIdx.x & 31;
    const int wave = threadIdx.x >> 5;
    const int j = blockIdx.x * (blockDim.x >> 5) + wave;   // 0..1023

    const __bf16* wr = Whh + (size_t)j * HD;
    const __bf16* wz = wr + (size_t)HD * HD;
    const __bf16* wn = wz + (size_t)HD * HD;

    float ar = 0.0f, az = 0.0f, an = 0.0f;
    #pragma unroll 4
    for (int k = lane; k < HD; k += 32) {
        float hv = h_prev[k];
        ar += (float)wr[k] * hv;
        az += (float)wz[k] * hv;
        an += (float)wn[k] * hv;
    }
    #pragma unroll
    for (int off = 16; off > 0; off >>= 1) {
        ar += __shfl_xor(ar, off, 32);
        az += __shfl_xor(az, off, 32);
        an += __shfl_xor(an, off, 32);
    }
    if (lane == 0) {
        float gr = ar + bhh[j];
        float gz = az + bhh[HD + j];
        float gn = an + bhh[2 * HD + j];
        float ir = gi_t[j], iz = gi_t[HD + j], inn = gi_t[2 * HD + j];
        float r = 1.0f / (1.0f + __expf(-(ir + gr)));
        float z = 1.0f / (1.0f + __expf(-(iz + gz)));
        float n = tanhf(inn + r * gn);
        float hp = h_prev[j];
        float hn = (1.0f - z) * n + z * hp;
        h_next[j] = hn;
        out_t[j]  = hn;
    }
}

// ---------------- Kernel 6: write h_final ----------------
__global__ void copy_final(const float* __restrict__ h, float* __restrict__ dst) {
    int i = blockIdx.x * blockDim.x + threadIdx.x;
    if (i < HD) dst[i] = h[i];
}

extern "C" void kernel_launch(void* const* d_in, const int* in_sizes, int n_in,
                              void* d_out, int out_size, void* d_ws, size_t ws_size,
                              hipStream_t stream) {
    const int*   tokens = (const int*)  d_in[0];
    const float* emb    = (const float*)d_in[1];
    const float* W_ih   = (const float*)d_in[2];
    const float* W_hh   = (const float*)d_in[3];
    const float* b_ih   = (const float*)d_in[4];
    const float* b_hh   = (const float*)d_in[5];
    float* out = (float*)d_out;

    // Workspace carve (all 16B aligned)
    char* ws = (char*)d_ws;
    __bf16* Xb   = (__bf16*)(ws);                          //  8,388,608 B
    __bf16* Wib  = (__bf16*)(ws + 8388608);                //  6,291,456 B
    __bf16* Whb  = (__bf16*)(ws + 14680064);               //  6,291,456 B
    float*  gi   = (float*) (ws + 20971520);               // 50,331,648 B
    float*  hbuf = (float*) (ws + 71303168);               //      8,192 B

    // 1) embedding gather -> bf16
    embed_to_bf16<<<SEQL, 256, 0, stream>>>(tokens, emb, Xb);
    // 2) weights -> bf16
    const int NW = H3 * HD;
    conv_to_bf16<<<NW / 256, 256, 0, stream>>>(W_ih, Wib, NW);
    conv_to_bf16<<<NW / 256, 256, 0, stream>>>(W_hh, Whb, NW);
    // 3) gi = X @ W_ih^T + b_ih   (WMMA, LDS-staged, async double-buffered)
    gemm_gi_wmma<<<dim3(H3 / 128, SEQL / 128), 256, 0, stream>>>(Xb, Wib, b_ih, gi);
    // 4) h = 0
    zero_h<<<8, 256, 0, stream>>>(hbuf);
    // 5) sequential scan: one launch per timestep (stream order = dependency)
    for (int t = 0; t < SEQL; ++t) {
        const float* hp = hbuf + (t & 1) * HD;
        float*       hn = hbuf + ((t + 1) & 1) * HD;
        gru_step<<<HD / 8, 256, 0, stream>>>(Whb, b_hh, gi + (size_t)t * H3,
                                             hp, hn, out + (size_t)t * HD);
    }
    // 6) h_final (after t=4095, result lives in buffer 0)
    copy_final<<<4, 256, 0, stream>>>(hbuf, out + (size_t)SEQL * HD);
}
